// DirectionAlignmentLoss_13357348290857
// MI455X (gfx1250) — matrix operand order repaired and checked by reading, not compiled, not run on hardware
//
#include <hip/hip_runtime.h>

typedef _Float16 half8 __attribute__((ext_vector_type(8)));
typedef _Float16 v16h  __attribute__((ext_vector_type(16)));
typedef float    v8f   __attribute__((ext_vector_type(8)));

#define DDIM 256
#define CCLS 64

__device__ __forceinline__ float waveReduceSum(float v) {
#pragma unroll
  for (int off = 1; off < 32; off <<= 1) v += __shfl_xor(v, off, 32);
  return v;
}

__device__ __forceinline__ void atomicMaxF(float* addr, float val) {
  if (val >= 0.0f) atomicMax((int*)addr, __float_as_int(val));
  else             atomicMin((unsigned int*)addr, __float_as_uint(val));
}

// ---------------------------------------------------------------- init
__global__ void init_k(float* hardest, float* wrongsum, float* align_sum, int B) {
  int i = blockIdx.x * blockDim.x + threadIdx.x;
  if (i < B) ((int*)hardest)[i] = (int)0xFF800000;  // -inf
  if (i < CCLS) wrongsum[i] = 0.0f;
  if (i == 0) *align_sum = 0.0f;
}

// ------------------------------------------------- row normalize -> f16
__global__ void normalize_k(const float* __restrict__ dirs,
                            _Float16* __restrict__ dnh, int B) {
  int wid  = (blockIdx.x * blockDim.x + threadIdx.x) >> 5;
  int lane = threadIdx.x & 31;
  if (wid >= B) return;
  const float* row = dirs + (size_t)wid * DDIM + lane * 8;
  float f[8];
#pragma unroll
  for (int j = 0; j < 8; ++j) f[j] = row[j];
  float s = 0.0f;
#pragma unroll
  for (int j = 0; j < 8; ++j) s += f[j] * f[j];
  s = waveReduceSum(s);
  float inv = 1.0f / fmaxf(sqrtf(s), 1e-12f);
  half8 h;
#pragma unroll
  for (int j = 0; j < 8; ++j) h[j] = (_Float16)(f[j] * inv);
  *(half8*)(dnh + (size_t)wid * DDIM + lane * 8) = h;
}

// ------------------------------------------------- per-class sums/counts
__global__ void class_sums_k(const _Float16* __restrict__ dnh,
                             const int* __restrict__ labels,
                             float* __restrict__ sums, int* __restrict__ counts, int B) {
  int c = blockIdx.x, d = threadIdx.x;
  float acc = 0.0f;
  int cnt = 0;
  for (int r = 0; r < B; ++r) {
    if (labels[r] == c) { acc += (float)dnh[(size_t)r * DDIM + d]; ++cnt; }
  }
  sums[c * DDIM + d] = acc;
  if (d == 0) counts[c] = cnt;
}

// ------------------------------------------------- prototypes (normalize)
__global__ void protos_k(const float* __restrict__ sums,
                         const float* __restrict__ class_protos,
                         const int* __restrict__ counts,
                         float* __restrict__ protos) {
  __shared__ float red[DDIM];
  int c = blockIdx.x, d = threadIdx.x;
  float s = sums[c * DDIM + d];
  float p = class_protos[c * DDIM + d];

  red[d] = s * s; __syncthreads();
  for (int off = DDIM / 2; off > 0; off >>= 1) {
    if (d < off) red[d] += red[d + off];
    __syncthreads();
  }
  float n2s = red[0]; __syncthreads();

  red[d] = p * p; __syncthreads();
  for (int off = DDIM / 2; off > 0; off >>= 1) {
    if (d < off) red[d] += red[d + off];
    __syncthreads();
  }
  float n2p = red[0];

  float out = (counts[c] > 0) ? s / fmaxf(sqrtf(n2s), 1e-12f)
                              : p / fmaxf(sqrtf(n2p), 1e-12f);
  protos[c * DDIM + d] = out;
}

// --------------------------------- alignment + separation (wave per row)
__global__ void align_sep_k(const _Float16* __restrict__ dnh,
                            const int* __restrict__ labels,
                            const float* __restrict__ protos,
                            float* __restrict__ wrongsum,
                            float* __restrict__ align_sum, int B) {
  __shared__ float wrongL[CCLS];
  __shared__ float alignL;
  int tid = threadIdx.x;
  if (tid < CCLS) wrongL[tid] = 0.0f;
  if (tid == 0) alignL = 0.0f;
  __syncthreads();

  int wid = tid >> 5, lane = tid & 31;
  int row = blockIdx.x * 8 + wid;

  half8 h = *(const half8*)(dnh + (size_t)row * DDIM + lane * 8);
  float f[8];
#pragma unroll
  for (int j = 0; j < 8; ++j) f[j] = (float)h[j];
  int labRow = labels[row];

  float alignAcc = 0.0f;
  for (int c = 0; c < CCLS; ++c) {
    const float* pp = protos + c * DDIM + lane * 8;
    float part = 0.0f;
#pragma unroll
    for (int j = 0; j < 8; ++j) part += f[j] * pp[j];
    part = waveReduceSum(part);
    if (lane == 0) {
      if (c == labRow) {
        alignAcc += 1.0f - part;
      } else {
        float w = fmaxf(part - 0.2f, 0.0f);
        if (w > 0.0f) atomicAdd(&wrongL[c], w);
      }
    }
  }
  if (lane == 0) atomicAdd(&alignL, alignAcc);
  __syncthreads();
  if (tid < CCLS) {
    float w = wrongL[tid];
    if (w != 0.0f) atomicAdd(&wrongsum[tid], w);
  }
  if (tid == 0) atomicAdd(align_sum, alignL);
}

// --------------------------------- hard-negative: WMMA Gram-matrix tiles
// Block = 256 threads (8 waves), tile = 128 rows x 128 cols, K = 256.
// Per-lane column max is kept in registers; the 16-lane cross-reduce is
// deferred until after the column loop (one ds_bpermute pass per block-wave
// instead of one per subtile).
__global__ void __launch_bounds__(256) hard_k(const _Float16* __restrict__ dnh,
                                              const int* __restrict__ labels,
                                              float* __restrict__ hardest, int B) {
  int tid  = threadIdx.x;
  int wid  = tid >> 5;
  int lane = tid & 31;
  int hi   = lane >> 4;   // half-wave index
  int ln   = lane & 15;

  int rowbase = blockIdx.y * 128 + wid * 16;
  int colbase = blockIdx.x * 128;

  int labR[8];
#pragma unroll
  for (int v = 0; v < 8; ++v) labR[v] = labels[rowbase + v + hi * 8];

  // Preload A fragments for all 8 K-chunks (rows fixed per wave, reused
  // across the 8 column sub-tiles). A layout: M = lane%16; lanes 0-15 hold
  // K = {0..7, 16..23}, lanes 16-31 hold K = {8..15, 24..31} per 32-chunk.
  v16h afrag[8];
  const _Float16* pa = dnh + (size_t)(rowbase + ln) * DDIM;
#pragma unroll
  for (int kc = 0; kc < 8; ++kc) {
    int kb = kc * 32;
    half8 lo  = *(const half8*)(pa + kb + hi * 8);
    half8 hi8 = *(const half8*)(pa + kb + 16 + hi * 8);
    afrag[kc] = __builtin_shufflevector(lo, hi8, 0, 1, 2, 3, 4, 5, 6, 7,
                                        8, 9, 10, 11, 12, 13, 14, 15);
  }

  // Per-lane running max over the columns this lane holds (N = ln).
  float rowmax[8];
#pragma unroll
  for (int v = 0; v < 8; ++v) rowmax[v] = -__builtin_inff();

#pragma unroll 2
  for (int cs = 0; cs < 8; ++cs) {
    int col = colbase + cs * 16 + ln;
    const _Float16* pb = dnh + (size_t)col * DDIM;
    int labC = labels[col];

    v8f acc = {};
#pragma unroll
    for (int kc = 0; kc < 8; ++kc) {
      // B layout (K x N, N = lane%16): lanes 0-15 hold K={0..7,16..23} of the
      // 32-chunk, lanes 16-31 hold K={8..15,24..31}. B[k][n] = dn[col_n][k].
      int kb = kc * 32 + hi * 8;
      half8 b0 = *(const half8*)(pb + kb);
      half8 b1 = *(const half8*)(pb + kb + 16);
      v16h bfrag = __builtin_shufflevector(b0, b1, 0, 1, 2, 3, 4, 5, 6, 7,
                                           8, 9, 10, 11, 12, 13, 14, 15);
      acc = __builtin_amdgcn_wmma_f32_16x16x32_f16(
          false, afrag[kc], false, bfrag, (short)0, acc, false, false);
    }

    // C/D layout: lane ln = column N, VGPR v = row (v + 8*hi).
    // Mask same-label pairs, keep a per-lane max (no cross-lane traffic).
#pragma unroll
    for (int v = 0; v < 8; ++v) {
      float val = (labR[v] == labC) ? -__builtin_inff() : acc[v];
      rowmax[v] = fmaxf(rowmax[v], val);
    }
  }

  // One deferred 16-lane max reduce per row register (xor offsets 1..8
  // stay within each 16-lane half, which holds a fixed row set).
#pragma unroll
  for (int v = 0; v < 8; ++v) {
#pragma unroll
    for (int off = 1; off < 16; off <<= 1)
      rowmax[v] = fmaxf(rowmax[v], __shfl_xor(rowmax[v], off, 32));
    if (ln == v) atomicMaxF(&hardest[rowbase + v + hi * 8], rowmax[v]);
  }
}

// ---------------------------------------------------------------- final
__global__ void finalize_k(const float* __restrict__ hardest,
                           const float* __restrict__ wrongsum,
                           const int* __restrict__ counts,
                           const float* __restrict__ align_sum,
                           float* __restrict__ out, int B) {
  __shared__ float red[256];
  int tid = threadIdx.x;
  float hs = 0.0f;
  for (int i = tid; i < B; i += 256) {
    int bits = ((const int*)hardest)[i];
    if (bits != (int)0xFF800000) {  // has_neg
      hs += fmaxf(__int_as_float(bits) - 0.3f, 0.0f);
    }
  }
  red[tid] = hs; __syncthreads();
  for (int off = 128; off > 0; off >>= 1) {
    if (tid < off) red[tid] += red[tid + off];
    __syncthreads();
  }
  if (tid == 0) {
    float l_hard = red[0] / (float)B;
    float sep = 0.0f;
    for (int c = 0; c < CCLS; ++c) {
      int neg = B - counts[c];
      sep += (neg > 0) ? wrongsum[c] / (float)neg : 0.0f;
    }
    float l_sep   = sep / (float)CCLS;
    float l_align = align_sum[0] / (float)B;
    out[0] = 0.15f * l_align + 0.1f * l_sep + 0.05f * l_hard;
  }
}

extern "C" void kernel_launch(void* const* d_in, const int* in_sizes, int n_in,
                              void* d_out, int out_size, void* d_ws, size_t ws_size,
                              hipStream_t stream) {
  const float* dirs         = (const float*)d_in[0];
  const int*   labels       = (const int*)d_in[1];
  const float* class_protos = (const float*)d_in[2];
  float* out = (float*)d_out;

  const int B = in_sizes[0] / DDIM;  // 8192

  char* ws = (char*)d_ws;
  size_t off = 0;
  _Float16* dnh    = (_Float16*)(ws + off); off += (size_t)B * DDIM * sizeof(_Float16);
  float* sums      = (float*)(ws + off);    off += (size_t)CCLS * DDIM * sizeof(float);
  float* protos    = (float*)(ws + off);    off += (size_t)CCLS * DDIM * sizeof(float);
  int*   counts    = (int*)(ws + off);      off += 256;
  float* wrongsum  = (float*)(ws + off);    off += 256;
  float* align_sum = (float*)(ws + off);    off += 256;
  float* hardest   = (float*)(ws + off);    off += (size_t)B * sizeof(float);

  init_k<<<(B + 255) / 256, 256, 0, stream>>>(hardest, wrongsum, align_sum, B);
  normalize_k<<<B / 8, 256, 0, stream>>>(dirs, dnh, B);
  class_sums_k<<<CCLS, DDIM, 0, stream>>>(dnh, labels, sums, counts, B);
  protos_k<<<CCLS, DDIM, 0, stream>>>(sums, class_protos, counts, protos);
  align_sep_k<<<B / 8, 256, 0, stream>>>(dnh, labels, protos, wrongsum, align_sum, B);

  dim3 grid(B / 128, B / 128);
  hard_k<<<grid, 256, 0, stream>>>(dnh, labels, hardest, B);

  finalize_k<<<1, 256, 0, stream>>>(hardest, wrongsum, counts, align_sum, out, B);
}
